// SimpleGATNode_53463752900657
// MI455X (gfx1250) — matrix operand order repaired and checked by reading, not compiled, not run on hardware
//
#include <hip/hip_runtime.h>
#include <hip/hip_bf16.h>

typedef __attribute__((ext_vector_type(16))) _Float16 v16h;
typedef __attribute__((ext_vector_type(8)))  _Float16 v8h;
typedef __attribute__((ext_vector_type(8)))  float    v8f;

#define NEG_SLOPE 0.2f

#define SHUF16(a, b) __builtin_shufflevector(a, b, 0,1,2,3,4,5,6,7,8,9,10,11,12,13,14,15)

// ---------------------------------------------------------------------------
// f32 -> f16 elementwise convert
__global__ __launch_bounds__(256) void k_cvt_f16(const float* __restrict__ src,
                                                 _Float16* __restrict__ dst, int n) {
    int i = blockIdx.x * 256 + threadIdx.x;
    if (i < n) dst[i] = (_Float16)src[i];
}

// W[K,Nout] f32 -> WT[Nout,K] f16 (transposed convert)
__global__ __launch_bounds__(256) void k_cvtT_f16(const float* __restrict__ W,
                                                  _Float16* __restrict__ WT,
                                                  int K, int Nout) {
    int i = blockIdx.x * 256 + threadIdx.x;
    if (i >= K * Nout) return;
    int k = i / Nout, n = i % Nout;
    WT[(size_t)n * K + k] = (_Float16)W[i];
}

__global__ __launch_bounds__(256) void k_zero_f32(float* __restrict__ p, int n) {
    int i = blockIdx.x * 256 + threadIdx.x;
    if (i < n) p[i] = 0.0f;
}

// ---------------------------------------------------------------------------
// WMMA GEMM: C[M,Nout] = A[M,K](f16) * B, with BT[Nout,K] = B transposed (f16).
// Each wave computes a 16 x (16*NT) strip, f32 accumulate. The k-loop is
// unrolled by 2 as a ping-pong over two disjoint fragment register sets, so
// the next k-step's loads are in flight during the WMMAs and no register
// copies are needed for buffer rotation. Requires K % 64 == 0.
// A-fragment (16-bit A 16x32): lanes 0-15 row M=l hold K 0..15,
// lanes 16-31 hold K 16..31. B-fragment: lane = column N, lanes 0-15
// hold K 0..15, lanes 16-31 K 16..31 (contiguous 32B in BT rows).
template <int NT>
__global__ __launch_bounds__(256) void k_gemm_wmma(const _Float16* __restrict__ A,
                                                   const _Float16* __restrict__ BT,
                                                   float* __restrict__ C,
                                                   int M, int K, int Nout) {
    const int wave = (blockIdx.x * 256 + threadIdx.x) >> 5;
    const int lane = threadIdx.x & 31;
    const int mTiles = M >> 4;
    const int tileM  = wave % mTiles;
    const int colGrp = wave / mTiles;
    const int n0 = colGrp * (16 * NT);
    if (n0 >= Nout) return;              // wave-uniform guard

    const int half_ = lane >> 4;         // 0: lanes 0-15, 1: lanes 16-31
    const int l16   = lane & 15;

    const _Float16* arow = A  + (size_t)(tileM * 16 + l16) * K + 16 * half_;
    const _Float16* brow = BT + (size_t)(n0 + l16) * K + 16 * half_;

    v8f acc[NT];
    #pragma unroll
    for (int t = 0; t < NT; ++t) acc[t] = (v8f){};

    v16h aCur, bCur[NT];
    // prologue: fragments for k = 0
    {
        v8h p0 = *(const v8h*)(arow);
        v8h p1 = *(const v8h*)(arow + 8);
        aCur = SHUF16(p0, p1);
        #pragma unroll
        for (int t = 0; t < NT; ++t)
            bCur[t] = *(const v16h*)(brow + (size_t)(16 * t) * K);
    }

    for (int k = 0; k < K; k += 64) {
        // fragments for k+32 (always in range: K % 64 == 0)
        v16h aNxt, bNxt[NT];
        {
            v8h p0 = *(const v8h*)(arow + k + 32);
            v8h p1 = *(const v8h*)(arow + k + 40);
            aNxt = SHUF16(p0, p1);
            #pragma unroll
            for (int t = 0; t < NT; ++t)
                bNxt[t] = *(const v16h*)(brow + (size_t)(16 * t) * K + k + 32);
        }
        #pragma unroll
        for (int t = 0; t < NT; ++t)
            acc[t] = __builtin_amdgcn_wmma_f32_16x16x32_f16(false, aCur, false, bCur[t],
                                                            (short)0, acc[t], false, false);
        // fragments for k+64 -> directly into the Cur set (no copies)
        if (k + 64 < K) {                // uniform
            v8h p0 = *(const v8h*)(arow + k + 64);
            v8h p1 = *(const v8h*)(arow + k + 72);
            aCur = SHUF16(p0, p1);
            #pragma unroll
            for (int t = 0; t < NT; ++t)
                bCur[t] = *(const v16h*)(brow + (size_t)(16 * t) * K + k + 64);
        }
        #pragma unroll
        for (int t = 0; t < NT; ++t)
            acc[t] = __builtin_amdgcn_wmma_f32_16x16x32_f16(false, aNxt, false, bNxt[t],
                                                            (short)0, acc[t], false, false);
    }

    // C/D layout: VGPR i -> row tileM*16 + half_*8 + i, column n0 + t*16 + l16
    const int rbase = tileM * 16 + half_ * 8;
    #pragma unroll
    for (int t = 0; t < NT; ++t) {
        float* cb = C + (size_t)rbase * Nout + (n0 + t * 16 + l16);
        #pragma unroll
        for (int i = 0; i < 8; ++i) cb[(size_t)i * Nout] = acc[t][i];
    }
}

// ---------------------------------------------------------------------------
// alpha_src/alpha_dst per (node, head); also init max (ordered -inf) and denom
template <int HEADS, int CH>
__global__ __launch_bounds__(256) void k_alpha(const float* __restrict__ Hm,
                                               const float* __restrict__ a_src,
                                               const float* __restrict__ a_dst,
                                               float* __restrict__ asrc,
                                               float* __restrict__ adst,
                                               unsigned* __restrict__ mbuf,
                                               float* __restrict__ denom,
                                               int n_nodes) {
    int i = blockIdx.x * 256 + threadIdx.x;
    if (i >= n_nodes * HEADS) return;
    int h = i % HEADS, n = i / HEADS;
    const float* row = Hm + (size_t)n * HEADS * CH + (size_t)h * CH;
    const float* as = a_src + h * CH;
    const float* ad = a_dst + h * CH;
    float s1 = 0.f, s2 = 0.f;
    #pragma unroll 8
    for (int c = 0; c < CH; ++c) { float v = row[c]; s1 += v * as[c]; s2 += v * ad[c]; }
    asrc[i] = s1; adst[i] = s2; mbuf[i] = 0u; denom[i] = 0.f;
}

__device__ __forceinline__ unsigned ford(float f) {
    unsigned u = __float_as_uint(f);
    return (u & 0x80000000u) ? ~u : (u | 0x80000000u);
}
__device__ __forceinline__ float funord(unsigned o) {
    unsigned u = (o & 0x80000000u) ? (o & 0x7FFFFFFFu) : ~o;
    return __uint_as_float(u);
}
__device__ __forceinline__ void edge_sd(const int* ei, int E_, int e, int& s, int& d) {
    if (e < E_) { s = ei[e]; d = ei[E_ + e]; } else { s = e - E_; d = s; }
}

// pass 1: segment max of leaky_relu(asrc[src]+adst[dst]) into mbuf[dst]
template <int HEADS>
__global__ __launch_bounds__(256) void k_edge_max(const int* __restrict__ ei, int E_,
                                                  int n_nodes,
                                                  const float* __restrict__ asrc,
                                                  const float* __restrict__ adst,
                                                  unsigned* __restrict__ mbuf) {
    long long i = (long long)blockIdx.x * 256 + threadIdx.x;
    long long tot = (long long)(E_ + n_nodes) * HEADS;
    if (i >= tot) return;
    int h = (int)(i % HEADS); int e = (int)(i / HEADS);
    int s, d; edge_sd(ei, E_, e, s, d);
    float ee = asrc[(size_t)s * HEADS + h] + adst[(size_t)d * HEADS + h];
    ee = ee > 0.f ? ee : ee * NEG_SLOPE;
    atomicMax(&mbuf[(size_t)d * HEADS + h], ford(ee));
}

// pass 2: denom[dst] += exp(e - m[dst])
template <int HEADS>
__global__ __launch_bounds__(256) void k_edge_den(const int* __restrict__ ei, int E_,
                                                  int n_nodes,
                                                  const float* __restrict__ asrc,
                                                  const float* __restrict__ adst,
                                                  const unsigned* __restrict__ mbuf,
                                                  float* __restrict__ denom) {
    long long i = (long long)blockIdx.x * 256 + threadIdx.x;
    long long tot = (long long)(E_ + n_nodes) * HEADS;
    if (i >= tot) return;
    int h = (int)(i % HEADS); int e = (int)(i / HEADS);
    int s, d; edge_sd(ei, E_, e, s, d);
    float ee = asrc[(size_t)s * HEADS + h] + adst[(size_t)d * HEADS + h];
    ee = ee > 0.f ? ee : ee * NEG_SLOPE;
    float mf = funord(mbuf[(size_t)d * HEADS + h]);
    if (!isfinite(mf)) mf = 0.f;
    atomicAdd(&denom[(size_t)d * HEADS + h], __expf(ee - mf));
}

// pass 3: out[dst,h,:] += (exp(e-m)/denom) * H[src,h,:]; 2 channels per thread
template <int HEADS, int CH>
__global__ __launch_bounds__(256) void k_edge_scatter(const int* __restrict__ ei, int E_,
                                                      int n_nodes,
                                                      const float* __restrict__ asrc,
                                                      const float* __restrict__ adst,
                                                      const unsigned* __restrict__ mbuf,
                                                      const float* __restrict__ denom,
                                                      const float* __restrict__ Hm,
                                                      float* __restrict__ outb) {
    constexpr int PER_EDGE = HEADS * CH / 2;
    long long i = (long long)blockIdx.x * 256 + threadIdx.x;
    long long tot = (long long)(E_ + n_nodes) * PER_EDGE;
    if (i >= tot) return;
    int e  = (int)(i / PER_EDGE);
    int r  = (int)(i % PER_EDGE);
    int h  = r / (CH / 2);
    int c2 = (r % (CH / 2)) * 2;
    int s, d; edge_sd(ei, E_, e, s, d);
    float ee = asrc[(size_t)s * HEADS + h] + adst[(size_t)d * HEADS + h];
    ee = ee > 0.f ? ee : ee * NEG_SLOPE;
    float mf = funord(mbuf[(size_t)d * HEADS + h]);
    if (!isfinite(mf)) mf = 0.f;
    float den = denom[(size_t)d * HEADS + h];
    den = den > 1e-16f ? den : 1e-16f;
    float alpha = __expf(ee - mf) / den;
    const float* hs = Hm   + (size_t)s * HEADS * CH + (size_t)h * CH + c2;
    float*       od = outb + (size_t)d * HEADS * CH + (size_t)h * CH + c2;
    atomicAdd(&od[0], alpha * hs[0]);
    atomicAdd(&od[1], alpha * hs[1]);
}

// out = elu(in + bias)
__global__ __launch_bounds__(256) void k_bias_elu(const float* __restrict__ in,
                                                  const float* __restrict__ b,
                                                  float* __restrict__ out,
                                                  int n_nodes, int F) {
    int i = blockIdx.x * 256 + threadIdx.x;
    if (i >= n_nodes * F) return;
    float v = in[i] + b[i & (F - 1)];   // F is a power of two (512)
    out[i] = v > 0.f ? v : (__expf(v) - 1.0f);
}

// final: log_softmax(in + b) over 16 classes
__global__ __launch_bounds__(256) void k_logsoftmax16(const float* __restrict__ in,
                                                      const float* __restrict__ b,
                                                      float* __restrict__ out,
                                                      int n_nodes) {
    int n = blockIdx.x * 256 + threadIdx.x;
    if (n >= n_nodes) return;
    float v[16];
    float mx = -3.402823466e38f;
    #pragma unroll
    for (int o = 0; o < 16; ++o) { v[o] = in[(size_t)n * 16 + o] + b[o]; mx = fmaxf(mx, v[o]); }
    float s = 0.f;
    #pragma unroll
    for (int o = 0; o < 16; ++o) s += __expf(v[o] - mx);
    float ls = __logf(s) + mx;
    #pragma unroll
    for (int o = 0; o < 16; ++o) out[(size_t)n * 16 + o] = v[o] - ls;
}

// ---------------------------------------------------------------------------
static inline int cdiv(long long a, long long b) { return (int)((a + b - 1) / b); }

struct Ws {
    float *featA, *featB, *Hmat, *asrc, *adst, *denom;
    unsigned* mbuf;
    _Float16 *Xh, *WhT;
};

template <int HEADS, int CH>
static void run_layer(const float* featIn, const float* Win, const float* asW,
                      const float* adW, int K, int N, int E, const int* ei,
                      const Ws& w, hipStream_t stream) {
    constexpr int Nout = HEADS * CH;
    k_cvt_f16<<<cdiv((long long)N * K, 256), 256, 0, stream>>>(featIn, w.Xh, N * K);
    k_cvtT_f16<<<cdiv((long long)K * Nout, 256), 256, 0, stream>>>(Win, w.WhT, K, Nout);
    long long mTiles = N / 16;
    if (Nout % 64 == 0) {
        long long waves = mTiles * (Nout / 64);
        k_gemm_wmma<4><<<cdiv(waves * 32, 256), 256, 0, stream>>>(w.Xh, w.WhT, w.Hmat, N, K, Nout);
    } else {
        long long waves = mTiles * (Nout / 16);
        k_gemm_wmma<1><<<cdiv(waves * 32, 256), 256, 0, stream>>>(w.Xh, w.WhT, w.Hmat, N, K, Nout);
    }
    k_alpha<HEADS, CH><<<cdiv((long long)N * HEADS, 256), 256, 0, stream>>>(
        w.Hmat, asW, adW, w.asrc, w.adst, w.mbuf, w.denom, N);
    k_zero_f32<<<cdiv((long long)N * Nout, 256), 256, 0, stream>>>(w.featB, N * Nout);
    long long eh = (long long)(E + N) * HEADS;
    k_edge_max<HEADS><<<cdiv(eh, 256), 256, 0, stream>>>(ei, E, N, w.asrc, w.adst, w.mbuf);
    k_edge_den<HEADS><<<cdiv(eh, 256), 256, 0, stream>>>(ei, E, N, w.asrc, w.adst, w.mbuf, w.denom);
    long long sw = (long long)(E + N) * (HEADS * CH / 2);
    k_edge_scatter<HEADS, CH><<<cdiv(sw, 256), 256, 0, stream>>>(
        ei, E, N, w.asrc, w.adst, w.mbuf, w.denom, w.Hmat, w.featB);
}

extern "C" void kernel_launch(void* const* d_in, const int* in_sizes, int n_in,
                              void* d_out, int out_size, void* d_ws, size_t ws_size,
                              hipStream_t stream) {
    const float* x   = (const float*)d_in[0];
    const int*   ei  = (const int*)  d_in[1];
    const float* W1  = (const float*)d_in[2];
    const float* as1 = (const float*)d_in[3];
    const float* ad1 = (const float*)d_in[4];
    const float* b1  = (const float*)d_in[5];
    const float* W2  = (const float*)d_in[6];
    const float* as2 = (const float*)d_in[7];
    const float* ad2 = (const float*)d_in[8];
    const float* b2  = (const float*)d_in[9];
    const float* W3  = (const float*)d_in[10];
    const float* as3 = (const float*)d_in[11];
    const float* ad3 = (const float*)d_in[12];
    const float* b3  = (const float*)d_in[13];

    const int IN = 256, HID = 64, H = 8, OUT = 16, F = H * HID; // F = 512
    const int N  = in_sizes[0] / IN;     // 20000
    const int E  = in_sizes[1] / 2;      // 320000

    // workspace layout (bytes)
    char* ws = (char*)d_ws;
    size_t off = 0;
    auto carve = [&](size_t bytes) { void* p = ws + off; off += (bytes + 255) & ~(size_t)255; return p; };
    Ws w;
    w.featA = (float*)   carve((size_t)N * F * 4);
    w.featB = (float*)   carve((size_t)N * F * 4);
    w.Hmat  = (float*)   carve((size_t)N * F * 4);
    w.Xh    = (_Float16*)carve((size_t)N * F * 2);
    w.WhT   = (_Float16*)carve((size_t)F * F * 2);
    w.asrc  = (float*)   carve((size_t)N * H * 4);
    w.adst  = (float*)   carve((size_t)N * H * 4);
    w.mbuf  = (unsigned*)carve((size_t)N * H * 4);
    w.denom = (float*)   carve((size_t)N * H * 4);

    // Layer 1: [N,256] -> [N,512], 8 heads x 64, concat + bias + ELU
    run_layer<8, 64>(x, W1, as1, ad1, IN, N, E, ei, w, stream);
    k_bias_elu<<<cdiv((long long)N * F, 256), 256, 0, stream>>>(w.featB, b1, w.featA, N, F);

    // Layer 2: [N,512] -> [N,512]
    run_layer<8, 64>(w.featA, W2, as2, ad2, F, N, E, ei, w, stream);
    k_bias_elu<<<cdiv((long long)N * F, 256), 256, 0, stream>>>(w.featB, b2, w.featA, N, F);

    // Layer 3: [N,512] -> [N,16], 1 head, mean==identity, + bias, log_softmax
    run_layer<1, 16>(w.featA, W3, as3, ad3, F, N, E, ei, w, stream);
    k_logsoftmax16<<<cdiv((long long)N, 256), 256, 0, stream>>>(w.featB, b3, (float*)d_out, N);
}
